// Asymm_3d_spconv_23854248362749
// MI455X (gfx1250) — matrix operand order repaired and checked by reading, not compiled, not run on hardware
//
#include <hip/hip_runtime.h>
#include <hip/hip_bf16.h>

#define N_VOX   200000
#define C_IN    128
#define C_OUTC  16
#define GZ      480
#define GY      360
#define GX      32
#define NBATCH  2
#define NTILES  (N_VOX / 16)     // 12500 exact
#define EPSF    1e-5f
#define SLOPE   0.01f

typedef __attribute__((ext_vector_type(16))) __bf16 v16bf;
typedef __attribute__((ext_vector_type(8)))  __bf16 v8bf;
typedef __attribute__((ext_vector_type(8)))  float  v8f;

union AK { v16bf v; v8bf h[2]; };
struct AFrag128 { AK k[4]; };    // one tap's 16x32 A fragments for all 4 k-steps (32 VGPRs)

// ---------------- grid build ----------------
__global__ void k_init(int* __restrict__ grid_idx, float* __restrict__ stats, int ngrid) {
    int i = blockIdx.x * 256 + threadIdx.x;
    if (i < ngrid) grid_idx[i] = -1;
    if (i < 128)   stats[i] = 0.f;
}

__global__ void k_scatter(const int* __restrict__ coords, int* __restrict__ grid_idx) {
    int i = blockIdx.x * 256 + threadIdx.x;
    if (i < N_VOX) {
        int b = coords[4*i+0], z = coords[4*i+1], y = coords[4*i+2], x = coords[4*i+3];
        grid_idx[((b*GZ + z)*GY + y)*GX + x] = i;
    }
}

// nbA: (1,3,3) pattern offsets (0,dy,dx);  nbB: (3,1,3) pattern offsets (dz,0,dx)
__global__ void k_build_nb(const int* __restrict__ coords, const int* __restrict__ grid_idx,
                           int* __restrict__ nbA, int* __restrict__ nbB) {
    int i = blockIdx.x * 256 + threadIdx.x;
    if (i >= N_VOX) return;
    int b = coords[4*i+0], z = coords[4*i+1], y = coords[4*i+2], x = coords[4*i+3];
    #pragma unroll
    for (int o = 0; o < 9; ++o) {
        int d0 = o / 3 - 1, d1 = o % 3 - 1;
        int ya = y + d0, xa = x + d1;
        int na = -1;
        if (ya >= 0 && ya < GY && xa >= 0 && xa < GX)
            na = grid_idx[((b*GZ + z)*GY + ya)*GX + xa];
        nbA[i*9 + o] = na;
        int zb = z + d0, xb = x + d1;
        int nb_ = -1;
        if (zb >= 0 && zb < GZ && xb >= 0 && xb < GX)
            nb_ = grid_idx[((b*GZ + zb)*GY + y)*GX + xb];
        nbB[i*9 + o] = nb_;
    }
}

// ---------------- feature cvt f32 -> bf16 ----------------
__global__ void k_cvt(const float* __restrict__ f, __bf16* __restrict__ h, int n) {
    int i = blockIdx.x * 256 + threadIdx.x;
    if (i < n) h[i] = (__bf16)f[i];
}

// ---------------- weight repack into per-lane B layout ----------------
// B fragment: lane holds column n = lane&15; K = (lane>>4)*8 + j for j<8, 16 + (lane>>4)*8 + (j-8) for j>=8
__global__ void k_repack128(const float* __restrict__ W, __bf16* __restrict__ dst) {
    int idx = blockIdx.x * 256 + threadIdx.x;          // [(o*4+ks)*32 + lane]*16 + j
    if (idx < 36*32*16) {
        int j    = idx & 15;
        int lane = (idx >> 4) & 31;
        int ks   = (idx >> 9) & 3;
        int o    = idx >> 11;
        int hi   = lane >> 4, n = lane & 15;
        int k    = ks*32 + ((j < 8) ? hi*8 + j : 16 + hi*8 + (j - 8));
        dst[idx] = (__bf16)W[(o*C_IN + k)*C_OUTC + n];
    }
}

// two taps fused per WMMA: pair p covers taps (2p, 2p+1); tap 9 (p=4 hi half) = zeros
__global__ void k_repack16(const float* __restrict__ W, __bf16* __restrict__ dst) {
    int idx = blockIdx.x * 256 + threadIdx.x;          // [p*32 + lane]*16 + j
    if (idx < 5*32*16) {
        int j    = idx & 15;
        int lane = (idx >> 4) & 31;
        int p    = idx >> 9;
        int hi   = lane >> 4, n = lane & 15;
        int K    = (j < 8) ? hi*8 + j : 16 + hi*8 + (j - 8);
        int o    = 2*p + (K >> 4);
        int kk   = K & 15;
        float v  = (o < 9) ? W[(o*16 + kk)*16 + n] : 0.f;
        dst[idx] = (__bf16)v;
    }
}

// load one tap's A fragments (4 k-steps) for this lane; masked lanes get zeros
__device__ __forceinline__ void loadA128(AFrag128& A, const __bf16* __restrict__ srcH,
                                         int nb, int hi, const v16bf& zero16) {
    if (nb >= 0) {
        const __bf16* base = srcH + (long)nb * C_IN + hi * 8;
        #pragma unroll
        for (int ks = 0; ks < 4; ++ks) {
            A.k[ks].h[0] = *(const v8bf*)(base + ks*32);        // K = ks*32 + hi*8 + 0..7
            A.k[ks].h[1] = *(const v8bf*)(base + ks*32 + 16);   // K = ks*32 + 16 + hi*8 + 0..7
        }
    } else {
        #pragma unroll
        for (int ks = 0; ks < 4; ++ks) A.k[ks].v = zero16;
    }
}

// ---------------- 128 -> 16 conv, 9 taps, WMMA bf16, 2-deep pipelined ----------------
__global__ __launch_bounds__(256) void k_conv128(
    const __bf16* __restrict__ srcH,   // [N][128]
    const int*    __restrict__ nbTab,  // [N][9]
    const __bf16* __restrict__ wpk,    // [36][32][16]
    float*        __restrict__ tOut,   // [N][16] pre-BN (post-lrelu)
    float*        __restrict__ stat)   // [32] sum / sumsq
{
    __shared__ alignas(16) __bf16 wl[36*32*16];
    __shared__ float ssum[16], ssq[16];
    {
        const uint4* s4 = (const uint4*)wpk;
        uint4*       d4 = (uint4*)wl;
        for (int i = threadIdx.x; i < (36*32*16)/8; i += 256) d4[i] = s4[i];
    }
    if (threadIdx.x < 16) { ssum[threadIdx.x] = 0.f; ssq[threadIdx.x] = 0.f; }
    __syncthreads();

    const int lane = threadIdx.x & 31;
    const int tile = blockIdx.x * 8 + (threadIdx.x >> 5);
    const int row  = lane & 15;
    const int hi   = lane >> 4;

    if (tile < NTILES) {
        v8f c0 = {}, c1 = {};
        v16bf zero16;
        #pragma unroll
        for (int j = 0; j < 16; ++j) zero16[j] = (__bf16)0.0f;

        // prefetch all neighbor indices for this voxel row
        const int* nbp = nbTab + ((long)tile*16 + row) * 9;
        int nbs[9];
        #pragma unroll
        for (int o = 0; o < 9; ++o) nbs[o] = nbp[o];

        // 2-deep software pipeline over taps: load tap o+1 while WMMAing tap o
        AFrag128 Abuf[2];
        loadA128(Abuf[0], srcH, nbs[0], hi, zero16);
        #pragma unroll
        for (int o = 0; o < 9; ++o) {
            AFrag128& cur = Abuf[o & 1];
            if (o < 8) loadA128(Abuf[(o + 1) & 1], srcH, nbs[o + 1], hi, zero16);
            // batch the 4 B-fragment LDS loads so one dscnt wait covers 4 WMMAs
            v16bf b[4];
            #pragma unroll
            for (int ks = 0; ks < 4; ++ks)
                b[ks] = *(const v16bf*)&wl[((o*4 + ks)*32 + lane)*16];
            #pragma unroll
            for (int ks = 0; ks < 4; ++ks) {
                if ((ks & 1) == 0)
                    c0 = __builtin_amdgcn_wmma_f32_16x16x32_bf16(false, cur.k[ks].v, false, b[ks],
                                                                 (short)0, c0, false, false);
                else
                    c1 = __builtin_amdgcn_wmma_f32_16x16x32_bf16(false, cur.k[ks].v, false, b[ks],
                                                                 (short)0, c1, false, false);
            }
        }
        float s = 0.f, q = 0.f;
        float* orow = tOut + ((long)tile*16 + hi*8)*C_OUTC + row;
        #pragma unroll
        for (int v = 0; v < 8; ++v) {
            float x = c0[v] + c1[v];
            x = (x >= 0.f) ? x : SLOPE * x;
            orow[(long)v * C_OUTC] = x;
            s += x; q += x * x;
        }
        atomicAdd(&ssum[row], s);
        atomicAdd(&ssq[row],  q);
    }
    __syncthreads();
    if (threadIdx.x < 16) {
        atomicAdd(&stat[threadIdx.x],      ssum[threadIdx.x]);
        atomicAdd(&stat[16 + threadIdx.x], ssq[threadIdx.x]);
    }
}

// ---------------- 16 -> 16 conv, 9 taps (2 per WMMA along K), pipelined ----------------
__global__ __launch_bounds__(256) void k_conv16(
    const __bf16* __restrict__ srcH,   // [N][16]
    const int*    __restrict__ nbTab,  // [N][9]
    const __bf16* __restrict__ wpk,    // [5][32][16]
    float*        __restrict__ tOut,
    float*        __restrict__ stat)
{
    __shared__ alignas(16) __bf16 wl[5*32*16];
    __shared__ float ssum[16], ssq[16];
    {
        const uint4* s4 = (const uint4*)wpk;
        uint4*       d4 = (uint4*)wl;
        for (int i = threadIdx.x; i < (5*32*16)/8; i += 256) d4[i] = s4[i];
    }
    if (threadIdx.x < 16) { ssum[threadIdx.x] = 0.f; ssq[threadIdx.x] = 0.f; }
    __syncthreads();

    const int lane = threadIdx.x & 31;
    const int tile = blockIdx.x * 8 + (threadIdx.x >> 5);
    const int row  = lane & 15;
    const int hi   = lane >> 4;

    if (tile < NTILES) {
        v8f c0 = {}, c1 = {};
        v8bf zero8;
        #pragma unroll
        for (int j = 0; j < 8; ++j) zero8[j] = (__bf16)0.0f;

        const int* nbp = nbTab + ((long)tile*16 + row) * 9;
        int nbs[9];
        #pragma unroll
        for (int o = 0; o < 9; ++o) nbs[o] = nbp[o];

        // pipeline: A for pair p+1 loads while pair p WMMAs
        AK Abuf[2];
        Abuf[0].h[0] = (nbs[0] >= 0) ? *(const v8bf*)(srcH + (long)nbs[0]*16 + hi*8) : zero8;
        Abuf[0].h[1] = (nbs[1] >= 0) ? *(const v8bf*)(srcH + (long)nbs[1]*16 + hi*8) : zero8;
        #pragma unroll
        for (int p = 0; p < 5; ++p) {
            AK& cur = Abuf[p & 1];
            if (p < 4) {
                AK& nxt = Abuf[(p + 1) & 1];
                int nb0 = nbs[2*p + 2];
                int nb1 = (p < 3) ? nbs[2*p + 3] : -1;
                nxt.h[0] = (nb0 >= 0) ? *(const v8bf*)(srcH + (long)nb0*16 + hi*8) : zero8;
                nxt.h[1] = (nb1 >= 0) ? *(const v8bf*)(srcH + (long)nb1*16 + hi*8) : zero8;
            }
            v16bf b = *(const v16bf*)&wl[(p*32 + lane)*16];
            if ((p & 1) == 0)
                c0 = __builtin_amdgcn_wmma_f32_16x16x32_bf16(false, cur.v, false, b,
                                                             (short)0, c0, false, false);
            else
                c1 = __builtin_amdgcn_wmma_f32_16x16x32_bf16(false, cur.v, false, b,
                                                             (short)0, c1, false, false);
        }
        float s = 0.f, q = 0.f;
        float* orow = tOut + ((long)tile*16 + hi*8)*C_OUTC + row;
        #pragma unroll
        for (int v = 0; v < 8; ++v) {
            float x = c0[v] + c1[v];
            x = (x >= 0.f) ? x : SLOPE * x;
            orow[(long)v * C_OUTC] = x;
            s += x; q += x * x;
        }
        atomicAdd(&ssum[row], s);
        atomicAdd(&ssq[row],  q);
    }
    __syncthreads();
    if (threadIdx.x < 16) {
        atomicAdd(&stat[threadIdx.x],      ssum[threadIdx.x]);
        atomicAdd(&stat[16 + threadIdx.x], ssq[threadIdx.x]);
    }
}

// ---------------- batchnorm apply ----------------
__global__ void k_bn_bf16(const float* __restrict__ t, const float* __restrict__ stat,
                          const float* __restrict__ g, const float* __restrict__ bb,
                          __bf16* __restrict__ outH) {
    int i = blockIdx.x * 256 + threadIdx.x;
    if (i < N_VOX * C_OUTC) {
        int   ch  = i & 15;
        float mu  = stat[ch] * (1.f / N_VOX);
        float var = stat[16 + ch] * (1.f / N_VOX) - mu * mu;
        float inv = rsqrtf(var + EPSF);
        outH[i] = (__bf16)((t[i] - mu) * inv * g[ch] + bb[ch]);
    }
}

__global__ void k_bn_f32(const float* __restrict__ t, const float* __restrict__ stat,
                         const float* __restrict__ g, const float* __restrict__ bb,
                         float* __restrict__ outF) {
    int i = blockIdx.x * 256 + threadIdx.x;
    if (i < N_VOX * C_OUTC) {
        int   ch  = i & 15;
        float mu  = stat[ch] * (1.f / N_VOX);
        float var = stat[16 + ch] * (1.f / N_VOX) - mu * mu;
        float inv = rsqrtf(var + EPSF);
        outF[i] = (t[i] - mu) * inv * g[ch] + bb[ch];
    }
}

__global__ void k_final(const float* __restrict__ t4, const float* __restrict__ stat,
                        const float* __restrict__ g, const float* __restrict__ bb,
                        const float* __restrict__ s2, float* __restrict__ out) {
    int i = blockIdx.x * 256 + threadIdx.x;
    if (i < N_VOX * C_OUTC) {
        int   ch  = i & 15;
        float mu  = stat[ch] * (1.f / N_VOX);
        float var = stat[16 + ch] * (1.f / N_VOX) - mu * mu;
        float inv = rsqrtf(var + EPSF);
        out[i] = (t4[i] - mu) * inv * g[ch] + bb[ch] + s2[i];
    }
}

// ---------------- host launcher ----------------
extern "C" void kernel_launch(void* const* d_in, const int* in_sizes, int n_in,
                              void* d_out, int out_size, void* d_ws, size_t ws_size,
                              hipStream_t stream) {
    const float* feats  = (const float*)d_in[0];
    const float* W1     = (const float*)d_in[1];
    const float* g0     = (const float*)d_in[2];
    const float* b0     = (const float*)d_in[3];
    const float* W1_2   = (const float*)d_in[4];
    const float* g0_2   = (const float*)d_in[5];
    const float* b0_2   = (const float*)d_in[6];
    const float* W2     = (const float*)d_in[7];
    const float* g1     = (const float*)d_in[8];
    const float* b1     = (const float*)d_in[9];
    const float* W3     = (const float*)d_in[10];
    const float* g2     = (const float*)d_in[11];
    const float* b2     = (const float*)d_in[12];
    const int*   coords = (const int*)d_in[13];

    char* ws  = (char*)d_ws;
    size_t off = 0;
    auto alloc = [&](size_t bytes) -> char* {
        char* p = ws + off;
        off += (bytes + 255) & ~(size_t)255;
        return p;
    };

    const int ngrid = NBATCH * GZ * GY * GX;         // 11,059,200
    int*    grid_idx = (int*)   alloc((size_t)ngrid * 4);
    int*    nbA      = (int*)   alloc((size_t)N_VOX * 9 * 4);
    int*    nbB      = (int*)   alloc((size_t)N_VOX * 9 * 4);
    __bf16* featsH   = (__bf16*)alloc((size_t)N_VOX * C_IN * 2);
    __bf16* s1H      = (__bf16*)alloc((size_t)N_VOX * C_OUTC * 2);
    __bf16* r1H      = (__bf16*)alloc((size_t)N_VOX * C_OUTC * 2);
    float*  tbuf     = (float*) alloc((size_t)N_VOX * C_OUTC * 4);
    float*  s2buf    = (float*) alloc((size_t)N_VOX * C_OUTC * 4);
    __bf16* wpk1     = (__bf16*)alloc(36*32*16*2);
    __bf16* wpk2     = (__bf16*)alloc(5*32*16*2);
    __bf16* wpk3     = (__bf16*)alloc(36*32*16*2);
    __bf16* wpk4     = (__bf16*)alloc(5*32*16*2);
    float*  stats    = (float*) alloc(4*32*4);

    // index structures + bf16 conversion + weight repack
    k_init    <<<(ngrid + 255)/256, 256, 0, stream>>>(grid_idx, stats, ngrid);
    k_scatter <<<(N_VOX + 255)/256, 256, 0, stream>>>(coords, grid_idx);
    k_build_nb<<<(N_VOX + 255)/256, 256, 0, stream>>>(coords, grid_idx, nbA, nbB);
    k_cvt     <<<(N_VOX*C_IN + 255)/256, 256, 0, stream>>>(feats, featsH, N_VOX*C_IN);
    k_repack128<<<72, 256, 0, stream>>>(W1, wpk1);
    k_repack128<<<72, 256, 0, stream>>>(W2, wpk3);
    k_repack16 <<<10, 256, 0, stream>>>(W1_2, wpk2);
    k_repack16 <<<10, 256, 0, stream>>>(W3, wpk4);

    const int cblocks = (NTILES + 7) / 8;             // 1563
    const int eblocks = (N_VOX * C_OUTC + 255) / 256; // 12500

    // s-branch: conv W1 (1,3,3) -> bn -> conv W1_2 (3,1,3) -> bn -> s2buf
    k_conv128<<<cblocks, 256, 0, stream>>>(featsH, nbA, wpk1, tbuf, stats + 0);
    k_bn_bf16<<<eblocks, 256, 0, stream>>>(tbuf, stats + 0, g0, b0, s1H);
    k_conv16 <<<cblocks, 256, 0, stream>>>(s1H, nbB, wpk2, tbuf, stats + 32);
    k_bn_f32 <<<eblocks, 256, 0, stream>>>(tbuf, stats + 32, g0_2, b0_2, s2buf);

    // r-branch: conv W2 (3,1,3) -> bn -> conv W3 (1,3,3) -> bn + add s
    k_conv128<<<cblocks, 256, 0, stream>>>(featsH, nbB, wpk3, tbuf, stats + 64);
    k_bn_bf16<<<eblocks, 256, 0, stream>>>(tbuf, stats + 64, g1, b1, r1H);
    k_conv16 <<<cblocks, 256, 0, stream>>>(r1H, nbA, wpk4, tbuf, stats + 96);
    k_final  <<<eblocks, 256, 0, stream>>>(tbuf, stats + 96, g2, b2, s2buf, (float*)d_out);
}
